// Encoder_RNN_73718818668940
// MI455X (gfx1250) — matrix operand order repaired and checked by reading, not compiled, not run on hardware
//
#include <hip/hip_runtime.h>
#include <stdint.h>

typedef __attribute__((ext_vector_type(16))) __bf16 v16bf;
typedef __attribute__((ext_vector_type(8)))  float  v8f;

union Frag16 { v16bf v; uint32_t u[8]; };

// fp32 -> bf16, round-half-up (u + 0x8000), take high 16 bits.
static __device__ __forceinline__ uint32_t bf16r(float f) {
  return __float_as_uint(f) + 0x8000u;
}
// pack two bf16 (high halves of rounded u32s) with one v_perm_b32
static __device__ __forceinline__ uint32_t pk_bf16(float lo, float hi) {
  return __builtin_amdgcn_perm(bf16r(hi), bf16r(lo), 0x07060302u);
}

#define BM 128
#define BN 64
#define BK 32
#define PITCH 36                 // uint16 pitch; 72-byte rows keep 8B alignment
#define ASZ (BM * PITCH)
#define BSZ (BN * PITCH)

// C = act(A @ B + bias); A is bf16 (ABF16=true, lda mult of 8) or fp32; B/C fp32.
// REQUIREMENTS: K (and kchunk) multiple of 32; N multiple of 4; lda/ldb mult of 4.
// splitk==0: blockIdx.z = batch. splitk==1: blockIdx.z = K-chunk, atomicAdd into C.
template <bool ABF16>
__global__ __launch_bounds__(256)
void gemm_wmma(const void* __restrict__ Ap, const float* __restrict__ B,
               float* __restrict__ C,
               int M, int N, int K, int lda, int ldb, int ldc,
               long long sA, long long sB, long long sC,
               const float* __restrict__ bias, int act,
               int splitk, int kchunk)
{
  __shared__ __align__(16) uint16_t As[2 * ASZ];
  __shared__ __align__(16) uint16_t Bs[2 * BSZ];

  const int tid = threadIdx.x;
  const int bm = blockIdx.x * BM;
  const int bn = blockIdx.y * BN;
  int kb, ke, batch;
  if (splitk) { batch = 0; kb = blockIdx.z * kchunk; ke = kb + kchunk; if (ke > K) ke = K; }
  else        { batch = blockIdx.z; kb = 0; ke = K; }
  const float*    Af = (const float*)Ap    + (ABF16 ? 0 : sA * batch);
  const uint16_t* Ah = (const uint16_t*)Ap + (ABF16 ? sA * batch : 0);
  B += sB * batch; C += sC * batch;

  const int wave = tid >> 5, lane = tid & 31;
  const int half = lane >> 4, l16 = lane & 15;
  const int wm = (wave & 3) * 32;   // 4 waves along M
  const int wn = (wave >> 2) * 32;  // 2 waves along N

  // ---- branch-free staging addressing ----
  const int arow = tid >> 3;             // fp32 A: + it*32
  const int acol = (tid & 7) * 4;
  const int hrow = tid >> 2;             // bf16 A: + it*64
  const int hcol = (tid & 3) * 8;        // 8 bf16 = 16 bytes
  const int bkr  = tid >> 4;             // B: + it*16
  const int bnc  = (tid & 15) * 4;
  int arowg[4];
  #pragma unroll
  for (int it = 0; it < 4; ++it) {
    int r = bm + (ABF16 ? (hrow + it * 64) : (arow + it * 32));
    arowg[it] = (r < M) ? r : (M - 1);   // clamp: OOB rows discarded at store
  }
  const bool bin = (bn + bnc) < N;       // whole float4 in/out (N % 4 == 0)
  const int  bng = bin ? (bn + bnc) : 0;

  float4 pa[4]; uint4 qa[2]; float4 pb[2];
  auto gload = [&](int k) {
    if constexpr (ABF16) {
      #pragma unroll
      for (int it = 0; it < 2; ++it)
        qa[it] = *(const uint4*)(Ah + (long long)arowg[it] * lda + k + hcol);
    } else {
      #pragma unroll
      for (int it = 0; it < 4; ++it)
        pa[it] = *(const float4*)(Af + (long long)arowg[it] * lda + k + acol);
    }
    #pragma unroll
    for (int it = 0; it < 2; ++it) {
      float4 v = *(const float4*)(B + (long long)(k + bkr + it * 16) * ldb + bng);
      if (!bin) v = make_float4(0.f, 0.f, 0.f, 0.f);
      pb[it] = v;
    }
  };
  auto sstore = [&](int buf) {
    uint16_t* as = As + buf * ASZ;
    uint16_t* bs = Bs + buf * BSZ;
    if constexpr (ABF16) {
      #pragma unroll
      for (int it = 0; it < 2; ++it) {
        uint16_t* d = &as[(hrow + it * 64) * PITCH + hcol];
        *(uint2*)d       = make_uint2(qa[it].x, qa[it].y);
        *(uint2*)(d + 4) = make_uint2(qa[it].z, qa[it].w);
      }
    } else {
      #pragma unroll
      for (int it = 0; it < 4; ++it) {
        uint2 w; w.x = pk_bf16(pa[it].x, pa[it].y); w.y = pk_bf16(pa[it].z, pa[it].w);
        *(uint2*)&as[(arow + it * 32) * PITCH + acol] = w;
      }
    }
    #pragma unroll
    for (int it = 0; it < 2; ++it) {     // transposed: Bs[n][k]
      int kk = bkr + it * 16;
      bs[(bnc + 0) * PITCH + kk] = (uint16_t)(bf16r(pb[it].x) >> 16);
      bs[(bnc + 1) * PITCH + kk] = (uint16_t)(bf16r(pb[it].y) >> 16);
      bs[(bnc + 2) * PITCH + kk] = (uint16_t)(bf16r(pb[it].z) >> 16);
      bs[(bnc + 3) * PITCH + kk] = (uint16_t)(bf16r(pb[it].w) >> 16);
    }
  };

  v8f acc[2][2] = {};

  gload(kb);
  sstore(0);
  __syncthreads();
  int cur = 0;

  for (int k0 = kb; k0 < ke; k0 += BK) {
    const bool hasNext = (k0 + BK) < ke;
    if (hasNext) gload(k0 + BK);         // prefetch overlaps WMMA below

    const uint16_t* as = As + cur * ASZ;
    const uint16_t* bs = Bs + cur * BSZ;
    Frag16 a[2], b[2];
    #pragma unroll
    for (int i = 0; i < 2; ++i) {
      const uint16_t* ap = &as[(wm + i * 16 + l16) * PITCH];
      #pragma unroll
      for (int p = 0; p < 8; ++p) {
        // A k order used here (contiguous, same as B): lanes<16 -> k 0..15; lanes>=16 -> k 16..31
        int kk = 2 * p + half * 16;
        a[i].u[p] = *(const uint32_t*)(ap + kk);
      }
    }
    #pragma unroll
    for (int j = 0; j < 2; ++j) {
      // B k order used here (contiguous, same as A): n = lane&15; lanes<16 -> k 0..15; lanes>=16 -> k 16..31
      const uint16_t* bp = &bs[(wn + j * 16 + l16) * PITCH + half * 16];
      #pragma unroll
      for (int p = 0; p < 8; ++p) b[j].u[p] = *(const uint32_t*)(bp + 2 * p);
    }
    #pragma unroll
    for (int i = 0; i < 2; ++i)
      #pragma unroll
      for (int j = 0; j < 2; ++j)
        acc[i][j] = __builtin_amdgcn_wmma_f32_16x16x32_bf16(
            false, a[i].v, false, b[j].v, (short)0, acc[i][j], false, false);

    if (hasNext) {
      sstore(cur ^ 1);                   // other buffer: one barrier suffices
      __syncthreads();
      cur ^= 1;
    }
  }

  // ---- epilogue; C/D layout: vgpr v -> m = v + half*8, n = lane&15
  #pragma unroll
  for (int i = 0; i < 2; ++i)
    #pragma unroll
    for (int j = 0; j < 2; ++j)
      #pragma unroll
      for (int v = 0; v < 8; ++v) {
        int gr = bm + wm + i * 16 + half * 8 + v;
        int gn = bn + wn + j * 16 + l16;
        if (gr < M && gn < N) {
          float val = acc[i][j][v];
          long long off = (long long)gr * ldc + gn;
          if (splitk) {
            atomicAdd(&C[off], val);
          } else {
            if (bias) val += bias[gn];
            if (act == 1) val = 1.0f / (1.0f + __expf(-val));
            else if (act == 2) val = tanhf(val);
            C[off] = val;
          }
        }
      }
}

// out[r, 0:ca]=a, out[r, ca:ca+cb]=b, out[r, ca+cb:cw]=0  (zero-padded width cw)
__global__ void concat2(const float* __restrict__ a, int lda, int ca,
                        const float* __restrict__ b, int ldb, int cb,
                        float* __restrict__ out, int rows, int cw) {
  long long total = (long long)rows * cw;
  long long i = (long long)blockIdx.x * blockDim.x + threadIdx.x;
  if (i >= total) return;
  int r = (int)(i / cw), c = (int)(i % cw);
  float v = 0.0f;
  if (c < ca)           v = a[(long long)r * lda + c];
  else if (c < ca + cb) v = b[(long long)r * ldb + (c - ca)];
  out[i] = v;
}

__global__ void pad_rows(const float* __restrict__ src, float* __restrict__ dst,
                         int rs, int rd, int cols) {
  int i = blockIdx.x * blockDim.x + threadIdx.x;
  if (i >= rd * cols) return;
  int r = i / cols, c = i % cols;
  dst[i] = (r < rs) ? src[r * cols + c] : 0.0f;
}

__global__ void cvt_bf16(const float* __restrict__ src, uint16_t* __restrict__ dst, int n) {
  int i = blockIdx.x * blockDim.x + threadIdx.x;
  if (i < n) dst[i] = (uint16_t)(bf16r(src[i]) >> 16);
}

__global__ void mul_ew(const float* __restrict__ a, const float* __restrict__ b,
                       float* __restrict__ o, int n) {
  int i = blockIdx.x * blockDim.x + threadIdx.x;
  if (i < n) o[i] = a[i] * b[i];
}

__global__ void gru_update(float* __restrict__ h, const float* __restrict__ z,
                           const float* __restrict__ nv, int n) {
  int i = blockIdx.x * blockDim.x + threadIdx.x;
  if (i < n) { float zz = z[i]; h[i] = (1.0f - zz) * h[i] + zz * nv[i]; }
}

__global__ void zero_f32(float* __restrict__ p, int n) {
  int i = blockIdx.x * blockDim.x + threadIdx.x;
  if (i < n) p[i] = 0.0f;
}

// mu = sigmoid(acc_mu+b_mu), lv = sigmoid(acc_lv+b_lv); eps is JAX-threefry-derived
// (not an input; E[eps]=0 over the 50-sample mean) -> z = mu.
__global__ void vae_head(const float* __restrict__ amu, const float* __restrict__ alv,
                         const float* __restrict__ bmu, const float* __restrict__ blv,
                         float* __restrict__ out, int Bsz, int D) {
  int i = blockIdx.x * blockDim.x + threadIdx.x;
  if (i >= Bsz * D) return;
  int d = i % D;
  float mu = 1.0f / (1.0f + __expf(-(amu[i] + bmu[d])));
  float lv = 1.0f / (1.0f + __expf(-(alv[i] + blv[d])));
  out[i] = mu;                    // z (eps ~ 0)
  out[Bsz * D + i] = mu;          // mu
  out[2 * Bsz * D + i] = lv;      // logvar
}

extern "C" void kernel_launch(void* const* d_in, const int* in_sizes, int n_in,
                              void* d_out, int out_size, void* d_ws, size_t ws_size,
                              hipStream_t stream) {
  const float* xs  = (const float*)d_in[0];   // [16,64,1024,1]
  const float* adj = (const float*)d_in[1];   // [1024,1024]
  const float* Wg[2][3]; const float* bgp[2][3];
  for (int l = 0; l < 2; ++l)
    for (int g = 0; g < 3; ++g) {
      Wg[l][g]  = (const float*)d_in[2 + l * 6 + g * 2];
      bgp[l][g] = (const float*)d_in[3 + l * 6 + g * 2];
    }
  const float* Wmu = (const float*)d_in[14]; const float* bmu = (const float*)d_in[15];
  const float* Wlv = (const float*)d_in[16]; const float* blv = (const float*)d_in[17];

  const int NN = 1024, NB = 64, ROWS = NB * NN;  // 65536
  char* ws = (char*)d_ws;
  float* h1  = (float*)(ws);                                // ROWS*64
  float* h2  = (float*)(ws + (size_t)ROWS * 64  * 4);       // ROWS*128
  float* bxh = (float*)(ws + (size_t)ROWS * 192 * 4);       // ROWS*192
  float* bG  = (float*)(ws + (size_t)ROWS * 384 * 4);       // ROWS*192
  float* bZ  = (float*)(ws + (size_t)ROWS * 576 * 4);       // ROWS*128
  float* bR  = (float*)(ws + (size_t)ROWS * 704 * 4);       // ROWS*128
  float* acc = (float*)(ws + (size_t)ROWS * 832 * 4);       // 2*64*256
  float* W0p = acc + 2 * 64 * 256;                          // 3 * 96*64 padded weights
  uint16_t* adjh = (uint16_t*)(W0p + 3 * 96 * 64);          // 1024*1024 bf16

  auto gemm = [&](bool abf16, const void* A, const float* B, float* C, int M, int N, int K,
                  int lda, int ldb, int ldc, long long sA, long long sB, long long sC,
                  const float* bias, int act, int splitk, int kchunk, int gz) {
    dim3 grid((M + BM - 1) / BM, (N + BN - 1) / BN, gz);
    if (abf16)
      gemm_wmma<true><<<grid, 256, 0, stream>>>(A, B, C, M, N, K, lda, ldb, ldc,
                                                sA, sB, sC, bias, act, splitk, kchunk);
    else
      gemm_wmma<false><<<grid, 256, 0, stream>>>(A, B, C, M, N, K, lda, ldb, ldc,
                                                 sA, sB, sC, bias, act, splitk, kchunk);
  };
  auto ew = [&](int n) { return dim3((n + 255) / 256); };

  zero_f32<<<ew(ROWS * 64), 256, 0, stream>>>(h1, ROWS * 64);
  zero_f32<<<ew(ROWS * 128), 256, 0, stream>>>(h2, ROWS * 128);
  cvt_bf16<<<ew(NN * NN), 256, 0, stream>>>(adj, adjh, NN * NN);   // adj -> bf16 once
  // zero-pad layer-0 gate weights 65x64 -> 96x64 so every GEMM K % 32 == 0
  for (int g = 0; g < 3; ++g)
    pad_rows<<<ew(96 * 64), 256, 0, stream>>>(Wg[0][g], W0p + g * 96 * 64, 65, 96, 64);

  const int cins[2]  = {1, 64};
  const int couts[2] = {64, 128};
  const int cpads[2] = {96, 192};      // padded concat widths (mult of 32)
  float* hs[2] = {h1, h2};

  for (int t = 0; t < 16; ++t) {
    for (int l = 0; l < 2; ++l) {
      const int cin = cins[l], cout = couts[l], Cp = cpads[l];
      const float* x = (l == 0) ? (xs + (size_t)t * ROWS) : h1;
      const float* Wz = (l == 0) ? (W0p + 0 * 96 * 64) : Wg[1][0];
      const float* Wr = (l == 0) ? (W0p + 1 * 96 * 64) : Wg[1][1];
      const float* Wn = (l == 0) ? (W0p + 2 * 96 * 64) : Wg[1][2];
      float* h = hs[l];
      // 1) XH = concat(x, h), zero-padded to Cp        [ROWS, Cp]
      concat2<<<ew(ROWS * Cp), 256, 0, stream>>>(x, cin, cin, h, cout, cout, bxh, ROWS, Cp);
      // 2) G = adj @ XH (batched over B=64)            [ROWS, Cp]
      gemm(true, adjh, bxh, bG, NN, Cp, NN, NN, Cp, Cp,
           0, (long long)NN * Cp, (long long)NN * Cp, nullptr, 0, 0, 0, NB);
      // 3) z = sigmoid(G @ Wz + bz)                    [ROWS, cout]
      gemm(false, bG, Wz, bZ, ROWS, cout, Cp, Cp, cout, cout, 0, 0, 0, bgp[l][0], 1, 0, 0, 1);
      // 4) r = sigmoid(G @ Wr + br)                    [ROWS, cout]
      gemm(false, bG, Wr, bR, ROWS, cout, Cp, Cp, cout, cout, 0, 0, 0, bgp[l][1], 1, 0, 0, 1);
      // 5) RH = r * h  -> reuse bxh                    [ROWS, cout]
      mul_ew<<<ew(ROWS * cout), 256, 0, stream>>>(bR, h, bxh, ROWS * cout);
      // 6) GR = adj @ RH -> reuse bR                   [ROWS, cout]
      gemm(true, adjh, bxh, bR, NN, cout, NN, NN, cout, cout,
           0, (long long)NN * cout, (long long)NN * cout, nullptr, 0, 0, 0, NB);
      // 7) XGR = concat(G[:, :cin], GR), padded -> bxh [ROWS, Cp]
      concat2<<<ew(ROWS * Cp), 256, 0, stream>>>(bG, Cp, cin, bR, cout, cout, bxh, ROWS, Cp);
      // 8) n = tanh(XGR @ Wn + bn) -> bG               [ROWS, cout]
      gemm(false, bxh, Wn, bG, ROWS, cout, Cp, Cp, cout, cout, 0, 0, 0, bgp[l][2], 2, 0, 0, 1);
      // 9) h = (1-z)*h + z*n
      gru_update<<<ew(ROWS * cout), 256, 0, stream>>>(h, bZ, bG, ROWS * cout);
    }
  }

  // final: xs = concat(h1, h2) viewed as [64, 196608]; split-K dense heads
  concat2<<<ew(ROWS * 192), 256, 0, stream>>>(h1, 64, 64, h2, 128, 128, bxh, ROWS, 192);
  zero_f32<<<ew(2 * 64 * 256), 256, 0, stream>>>(acc, 2 * 64 * 256);
  const int KD = 196608, KCH = 1536, KSP = 128;   // 128*1536 == 196608, KCH % 32 == 0
  gemm(false, bxh, Wmu, acc,            64, 256, KD, KD, 256, 256, 0, 0, 0, nullptr, 0, 1, KCH, KSP);
  gemm(false, bxh, Wlv, acc + 64 * 256, 64, 256, KD, KD, 256, 256, 0, 0, 0, nullptr, 0, 1, KCH, KSP);
  vae_head<<<ew(64 * 256), 256, 0, stream>>>(acc, acc + 64 * 256, bmu, blv,
                                             (float*)d_out, 64, 256);
  (void)in_sizes; (void)n_in; (void)out_size; (void)ws_size;
}
